// MambaBackbone_77721728188888
// MI455X (gfx1250) — matrix operand order repaired and checked by reading, not compiled, not run on hardware
//
#include <hip/hip_runtime.h>

// ---------------------------------------------------------------------------
// MI455X (gfx1250, wave32) Mamba vision backbone.
// GEMMs (conv-as-im2col, W_in, W_x, W_out) on V_WMMA_F32_16X16X32_BF16 with
// f32 accumulate; selective scan as register-resident VALU kernel.
// GEMM tile: BM=64 x BN=128 x BK=64, 8 waves, 8 WMMA / staging iteration.
// A tile staged by the Tensor Data Mover (TENSOR_LOAD_TO_LDS) when the
// toolchain exposes the builtin; manual b128 staging otherwise.
// ---------------------------------------------------------------------------

typedef __bf16 bf16_t;
typedef __attribute__((ext_vector_type(16))) __bf16 v16bf;
typedef __attribute__((ext_vector_type(8)))  float  v8f;
typedef __attribute__((ext_vector_type(4)))  unsigned u32x4;
typedef __attribute__((ext_vector_type(4)))  int      i32x4;
typedef __attribute__((ext_vector_type(8)))  int      i32x8;

#if defined(__AMDGCN__) && __has_builtin(__builtin_amdgcn_tensor_load_to_lds) && \
    __has_builtin(__builtin_amdgcn_s_wait_tensorcnt)
#define USE_TDM 1
#else
#define USE_TDM 0
#endif

static __device__ __forceinline__ float sigmoidf_(float x) {
  return 1.0f / (1.0f + __expf(-x));
}

// ---- f32 -> bf16 convert with K zero-padding (weights; Ksrc<=Kdst) ---------
__global__ void k_convert_pad(const float* __restrict__ w, bf16_t* __restrict__ o,
                              int rows, int Ksrc, int Kdst) {
  long idx = (long)blockIdx.x * blockDim.x + threadIdx.x;
  long total = (long)rows * Kdst;
  if (idx >= total) return;
  int r = (int)(idx / Kdst), k = (int)(idx % Kdst);
  o[idx] = (k < Ksrc) ? (bf16_t)w[(long)r * Ksrc + k] : (bf16_t)0.0f;
}

// ---- im2col (3x3, pad 1) into bf16 patches P[b][k][sp], k zero-padded ------
__global__ void k_im2col(const float* __restrict__ x, bf16_t* __restrict__ P,
                         int C, int H, int W, int OH, int OW, int stride, int Kd) {
  int b = blockIdx.z;
  long S = (long)OH * OW;
  long idx = (long)blockIdx.x * blockDim.x + threadIdx.x;
  if (idx >= (long)Kd * S) return;
  int k  = (int)(idx / S);
  int sp = (int)(idx % S);
  int oh = sp / OW, ow = sp % OW;
  float v = 0.0f;
  if (k < C * 9) {
    int ci = k / 9, r = k % 9, kh = r / 3, kw = r % 3;
    int ih = oh * stride + kh - 1, iw = ow * stride + kw - 1;
    if (ih >= 0 && ih < H && iw >= 0 && iw < W)
      v = x[((long)b * C + ci) * H * W + (long)ih * W + iw];
  }
  P[(long)b * Kd * S + (long)k * S + sp] = (bf16_t)v;
}

// ---- NCHW f32 -> sequence-major bf16 X[b][sp][c] ---------------------------
__global__ void k_to_seq(const float* __restrict__ f, bf16_t* __restrict__ X,
                         int C, long S) {
  int b = blockIdx.z;
  long idx = (long)blockIdx.x * blockDim.x + threadIdx.x;
  if (idx >= (long)C * S) return;
  int  c  = (int)(idx / S);
  long sp = idx % S;
  X[((long)b * S + sp) * C + c] = (bf16_t)f[((long)b * C + c) * S + sp];
}

// ---- causal depthwise conv1d (k=4) + SiLU; emits f32 and bf16 copies -------
__global__ void k_dwconv_silu(const float* __restrict__ xz, const float* __restrict__ cw,
                              const float* __restrict__ cb, float* __restrict__ xps,
                              bf16_t* __restrict__ xpb, int L, int di) {
  int b = blockIdx.z;
  long idx = (long)blockIdx.x * blockDim.x + threadIdx.x;
  if (idx >= (long)L * di) return;
  int l  = (int)(idx / di);
  int ch = (int)(idx % di);
  const float* base = xz + (long)b * L * (2 * di) + ch;  // xp = first di cols of xz
  float acc = cb[ch];
#pragma unroll
  for (int t = 0; t < 4; ++t) {
    int ls = l - 3 + t;                                  // left pad = 3
    if (ls >= 0) acc += cw[ch * 4 + t] * base[(long)ls * (2 * di)];
  }
  float s = acc * sigmoidf_(acc);
  long o = ((long)b * L + l) * di + ch;
  xps[o] = s;
  xpb[o] = (bf16_t)s;
}

// ---- selective scan, fused dt-projection + softplus + D skip + z-gate ------
__global__ void k_scan(const float* __restrict__ xdbl, const float* __restrict__ xps,
                       const float* __restrict__ xz,   const float* __restrict__ Wdt,
                       const float* __restrict__ bdt,  const float* __restrict__ Alog,
                       const float* __restrict__ Dp,   bf16_t* __restrict__ Yg,
                       int L, int di, int dtr, int G, int total) {
  int idx = blockIdx.x * blockDim.x + threadIdx.x;
  if (idx >= total) return;
  int b = idx / di, ch = idx % di;
  const float* xd = xdbl + (long)b * L * G;
  const float* xp = xps  + (long)b * L * di;
  const float* zz = xz   + (long)b * L * (2 * di);
  bf16_t*      yo = Yg   + (long)b * L * di;

  float Arow[16], h[16], wdt[16];
#pragma unroll
  for (int s = 0; s < 16; ++s) { Arow[s] = -__expf(Alog[ch * 16 + s]); h[s] = 0.0f; }
#pragma unroll
  for (int r = 0; r < 16; ++r) wdt[r] = (r < dtr) ? Wdt[r * di + ch] : 0.0f;
  const float bdtc = bdt[ch], Dc = Dp[ch];

  for (int l = 0; l < L; ++l) {
    const float* row = xd + (long)l * G;
    float dtl = bdtc;
#pragma unroll
    for (int r = 0; r < 16; ++r)
      if (r < dtr) dtl += row[r] * wdt[r];
    float dt  = (dtl > 20.0f) ? dtl : log1pf(__expf(dtl));   // softplus
    float xpv = xp[(long)l * di + ch];
    float y   = 0.0f;
#pragma unroll
    for (int s = 0; s < 16; ++s) {
      float Bm = row[dtr + s], Cm = row[dtr + 16 + s];
      h[s] = __expf(dt * Arow[s]) * h[s] + dt * Bm * xpv;    // h = dA*h + dBx
      y   += h[s] * Cm;
    }
    float zv = zz[(long)l * (2 * di) + di + ch];
    yo[(long)l * di + ch] = (bf16_t)((y + xpv * Dc) * (zv * sigmoidf_(zv)));
  }
}

// ---- bf16 WMMA GEMM: C[M][N] = A[M][K] @ B[K][N], f32 accumulate -----------
// EPI 0: plain f32 store. EPI 1: relu(C*scale[row]+bias[row]) (BN).
// EPI 2: transposed store C[col*ldTr + row] (seq -> NCHW).
// Requires K % 64 == 0 (host pads). M/N edges handled with hoisted guards
// (manual path) or hardware zero-fill via D#.tensor_dim1 (TDM path).
template <int EPI>
__global__ __launch_bounds__(256) void k_gemm_wmma(
    const bf16_t* __restrict__ A, const bf16_t* __restrict__ B, float* __restrict__ C,
    int M, int N, int K, long sA, long sB, long sC,
    const float* __restrict__ scale, const float* __restrict__ bias, int ldTr) {
  __shared__ __attribute__((aligned(16))) bf16_t As[64][72];    // [m][k] (+8 pad)
  __shared__ __attribute__((aligned(16))) bf16_t Bt[128][72];   // [n][k] transposed
  A += (long)blockIdx.z * sA;
  B += (long)blockIdx.z * sB;
  C += (long)blockIdx.z * sC;
  const int m0 = blockIdx.y * 64, n0 = blockIdx.x * 128;
  const int tid = threadIdx.x, lane = tid & 31, wave = tid >> 5;
  const int wm = (wave & 3) * 16;      // wave's 16-row group
  const int wn = (wave >> 2) * 64;     // wave's 64-col group
  const int fm    = lane & 15;
  const int kbase = (lane >> 4) * 8;   // unused; A frag also uses g16 (upper half-wave at k=+16)
  const int g16   = (lane >> 4) * 16;  // B frag: upper half-wave holds k=+16
  v8f acc[4] = {};

  const int ar = tid >> 2,  ac = (tid & 3) * 16;   // A staging: row, 16 k's
  const int bn = tid & 127, bk = (tid >> 7) * 32;  // B staging: col, 32 k's

  for (int k0 = 0; k0 < K; k0 += 64) {
#if USE_TDM
    if (wave == 0) {
      // TDM: 64x64 bf16 2D tile, row stride K elems, +16B LDS pad per row so it
      // lands exactly on As[64][72]. tensor_dim1 = M-m0 => HW zero-fill of edge.
      unsigned long long ga = (unsigned long long)(const void*)(A + (long)m0 * K + k0);
      unsigned lds_base =
          (unsigned)(unsigned long long)(__attribute__((address_space(3))) bf16_t*)&As[0][0];
      unsigned rowsRem = (unsigned)(M - m0); if (rowsRem > 65535u) rowsRem = 65535u;
      u32x4 g0; i32x8 g1;
      g0[0] = 1u;                                           // count=1, user, no gather
      g0[1] = lds_base;                                     // lds_addr
      g0[2] = (unsigned)(ga & 0xFFFFFFFFu);                 // global_addr lo
      g0[3] = (unsigned)((ga >> 32) & 0x01FFFFFFu) | (2u << 30);  // addr hi | type=2
      // data_size=1 (2B) | pad_enable | pad_interval=4 (32 DW) | pad_amount=3 (4 DW)
      g1[0] = (int)((1u << 16) | (1u << 20) | (4u << 22) | (3u << 25));
      g1[1] = (int)(64u << 16);                             // tensor_dim0 = 64 (lo16)
      g1[2] = (int)((rowsRem & 0xFFFFu) << 16);             // dim0 hi=0 | tensor_dim1 lo
      g1[3] = (int)(64u << 16);                             // dim1 hi=0 | tile_dim0 = 64
      g1[4] = (int)64;                                      // tile_dim1=64 | tile_dim2=0
      g1[5] = K;                                            // tensor_dim0_stride lo32
      g1[6] = 0; g1[7] = 0;                                 // stride hi | dim1_stride=0
      i32x4 gz = {0, 0, 0, 0};
#if defined(__clang_major__) && (__clang_major__ >= 23)
      i32x8 gz8 = {0, 0, 0, 0, 0, 0, 0, 0};
      __builtin_amdgcn_tensor_load_to_lds(g0, g1, gz, gz, gz8, 0);
#else
      __builtin_amdgcn_tensor_load_to_lds(g0, g1, gz, gz, 0);
#endif
    }
#else
    {  // ---- stage A tile: per-thread hoisted guard, two b128 copies ----
      int gm = m0 + ar;
      uint4* dst = (uint4*)&As[ar][ac];
      if (gm < M) {
        const uint4* src = (const uint4*)(A + (long)gm * K + k0 + ac);
        dst[0] = src[0];
        dst[1] = src[1];
      } else {
        uint4 z = {0u, 0u, 0u, 0u};
        dst[0] = z;
        dst[1] = z;
      }
    }
#endif
    {  // ---- stage B tile transposed via register buffer ----
      int gn = n0 + bn;
      unsigned tmp[16];
      if (gn < N) {
        const unsigned short* Bu = (const unsigned short*)B;
#pragma unroll
        for (int i = 0; i < 16; ++i) {
          unsigned lo = Bu[(long)(k0 + bk + 2 * i)     * N + gn];
          unsigned hi = Bu[(long)(k0 + bk + 2 * i + 1) * N + gn];
          tmp[i] = lo | (hi << 16);
        }
      } else {
#pragma unroll
        for (int i = 0; i < 16; ++i) tmp[i] = 0u;
      }
      uint4* dst = (uint4*)&Bt[bn][bk];
#pragma unroll
      for (int i = 0; i < 4; ++i) {
        uint4 q = {tmp[4 * i], tmp[4 * i + 1], tmp[4 * i + 2], tmp[4 * i + 3]};
        dst[i] = q;
      }
    }
#if USE_TDM
    if (wave == 0) __builtin_amdgcn_s_wait_tensorcnt(0);  // A tile landed in LDS
#endif
    __syncthreads();
    if (k0 + 64 < K) {  // prefetch next K slab (global_prefetch_b8)
      int pr = m0 + ar; if (pr >= M) pr = M - 1;
      __builtin_prefetch(A + (long)pr * K + (k0 + 64) + ac, 0, 1);
      int pn = n0 + bn; if (pn >= N) pn = N - 1;
      __builtin_prefetch(B + (long)(k0 + 64 + (tid >> 7)) * N + pn, 0, 1);
    }
    union Frag { v16bf v; __bf16 e[16]; };
#pragma unroll
    for (int ks = 0; ks < 64; ks += 32) {
      Frag fa;  // A 16x32: e[0..15] k=g16..g16+15 (contiguous, matches B)
      __builtin_memcpy(&fa.e[0], &As[wm + fm][ks + g16], 32);
#pragma unroll
      for (int sub = 0; sub < 4; ++sub) {  // one A frag reused for 4 B frags
        Frag fb;  // B 32x16: lane half selects k block, contiguous in Bt row
        __builtin_memcpy(&fb.e[0], &Bt[wn + sub * 16 + fm][ks + g16], 32);
        acc[sub] = __builtin_amdgcn_wmma_f32_16x16x32_bf16(
            false, fa.v, false, fb.v, (short)0, acc[sub], false, false);
      }
    }
    __syncthreads();
  }

  // C/D layout: VGPR r -> row r (lanes 0-15) / row 8+r (lanes 16-31)
  const int rbase = (lane >> 4) * 8;
#pragma unroll
  for (int sub = 0; sub < 4; ++sub) {
    int col = n0 + wn + sub * 16 + fm;
    if (col >= N) continue;
#pragma unroll
    for (int r = 0; r < 8; ++r) {
      int row = m0 + wm + rbase + r;
      if (row >= M) continue;
      float v = acc[sub][r];
      if (EPI == 1) v = fmaxf(v * scale[row] + bias[row], 0.0f);
      if (EPI == 2) C[(long)col * ldTr + row] = v;
      else          C[(long)row * N + col] = v;
    }
  }
}

// ---------------------------------------------------------------------------
extern "C" void kernel_launch(void* const* d_in, const int* in_sizes, int n_in,
                              void* d_out, int out_size, void* d_ws, size_t ws_size,
                              hipStream_t stream) {
  (void)in_sizes; (void)n_in; (void)out_size; (void)ws_size;
  const int  B = 2;
  const int  dims[3]  = {64, 128, 256};
  const int  Hin[3]   = {224, 112, 56};
  const int  CinT[3]  = {3, 64, 128};
  const long outOff[3] = {0, 1605632, 2408448};  // 2*64*112^2, +2*128*56^2
  const int  Gp = 64;                            // padded dt-proj output width

  char* ws = (char*)d_ws;
  // persistent cross-stage conv feature map (max = stage1: 2*64*112*112 f32)
  const size_t F2_BYTES = (((size_t)2 * 64 * 112 * 112 * sizeof(float)) + 255) & ~(size_t)255;
  float* F2 = (float*)ws;
  const float* stageIn = (const float*)d_in[0];

  auto cdiv = [](long a, long b) { return (unsigned)((a + b - 1) / b); };

  for (int s = 0; s < 3; ++s) {
    const int  d = dims[s], di = 2 * d, dtr = d / 16, G = dtr + 32;
    const int  H = Hin[s], C = CinT[s];
    const int  OH2 = H / 2;
    const long S1 = (long)H * H, S2 = (long)OH2 * OH2, L = S2;
    const int  K1 = ((C * 9 + 63) / 64) * 64;  // pad K to 64 (27->64 in stage 1)
    const int  K2 = d * 9;                     // 576/1152/2304, all %64==0

    const float* p[15];
    for (int i = 0; i < 15; ++i) p[i] = (const float*)d_in[1 + 15 * s + i];
    // 0:conv1_w 1:bn1_g 2:bn1_b 3:conv2_w 4:bn2_g 5:bn2_b 6:W_in 7:conv_w
    // 8:conv_b 9:W_x 10:W_dt 11:b_dt 12:A_log 13:D 14:W_out

    size_t off = F2_BYTES;
    auto alloc = [&](size_t bytes) -> char* {
      char* ptr = ws + off;
      off += (bytes + 255) & ~(size_t)255;
      return ptr;
    };
    bf16_t* wA1 = (bf16_t*)alloc((size_t)d * K1 * 2);
    bf16_t* P1  = (bf16_t*)alloc((size_t)B * K1 * S1 * 2);
    float*  F1  = (float*) alloc((size_t)B * d * S1 * 4);
    bf16_t* wA2 = (bf16_t*)alloc((size_t)d * K2 * 2);
    bf16_t* P2  = (bf16_t*)alloc((size_t)B * K2 * S2 * 2);
    bf16_t* Xs  = (bf16_t*)alloc((size_t)B * L * d * 2);
    bf16_t* Wi  = (bf16_t*)alloc((size_t)d * 4 * d * 2);
    float*  XZ  = (float*) alloc((size_t)B * L * 4 * d * 4);
    float*  XPS = (float*) alloc((size_t)B * L * di * 4);
    bf16_t* XPB = (bf16_t*)alloc((size_t)B * L * di * 2);
    bf16_t* Wx  = (bf16_t*)alloc((size_t)di * Gp * 2);
    float*  XD  = (float*) alloc((size_t)B * L * Gp * 4);
    bf16_t* YB  = (bf16_t*)alloc((size_t)B * L * di * 2);
    bf16_t* Wo  = (bf16_t*)alloc((size_t)di * d * 2);

    // ---- conv1 (stride 1) + BN + ReLU ----
    k_convert_pad<<<cdiv((long)d * K1, 256), 256, 0, stream>>>(p[0], wA1, d, C * 9, K1);
    k_im2col<<<dim3(cdiv((long)K1 * S1, 256), 1, B), 256, 0, stream>>>(
        stageIn, P1, C, H, H, H, H, 1, K1);
    k_gemm_wmma<1><<<dim3(cdiv(S1, 128), cdiv(d, 64), B), 256, 0, stream>>>(
        wA1, P1, F1, d, (int)S1, K1, 0L, (long)K1 * S1, (long)d * S1, p[1], p[2], 0);

    // ---- conv2 (stride 2) + BN + ReLU ----
    k_convert_pad<<<cdiv((long)d * K2, 256), 256, 0, stream>>>(p[3], wA2, d, d * 9, K2);
    k_im2col<<<dim3(cdiv((long)K2 * S2, 256), 1, B), 256, 0, stream>>>(
        F1, P2, d, H, H, OH2, OH2, 2, K2);
    k_gemm_wmma<1><<<dim3(cdiv(S2, 128), cdiv(d, 64), B), 256, 0, stream>>>(
        wA2, P2, F2, d, (int)S2, K2, 0L, (long)K2 * S2, (long)d * S2, p[4], p[5], 0);

    // ---- Mamba block ----
    k_to_seq<<<dim3(cdiv((long)d * S2, 256), 1, B), 256, 0, stream>>>(F2, Xs, d, S2);
    k_convert_pad<<<cdiv((long)d * 4 * d, 256), 256, 0, stream>>>(p[6], Wi, d, 4 * d, 4 * d);
    k_gemm_wmma<0><<<dim3(cdiv(4 * d, 128), cdiv(L, 64), B), 256, 0, stream>>>(
        Xs, Wi, XZ, (int)L, 4 * d, d, L * d, 0L, L * 4 * d, nullptr, nullptr, 0);

    k_dwconv_silu<<<dim3(cdiv(L * di, 256), 1, B), 256, 0, stream>>>(
        XZ, p[7], p[8], XPS, XPB, (int)L, di);

    k_convert_pad<<<cdiv((long)di * Gp, 256), 256, 0, stream>>>(p[9], Wx, di, G, Gp);
    k_gemm_wmma<0><<<dim3(cdiv(Gp, 128), cdiv(L, 64), B), 256, 0, stream>>>(
        XPB, Wx, XD, (int)L, Gp, di, L * di, 0L, L * Gp, nullptr, nullptr, 0);

    k_scan<<<cdiv((long)B * di, 128), 128, 0, stream>>>(
        XD, XPS, XZ, p[10], p[11], p[12], p[13], YB, (int)L, di, dtr, Gp, B * di);

    k_convert_pad<<<cdiv((long)di * d, 256), 256, 0, stream>>>(p[14], Wo, di, d, d);
    float* outChunk = (float*)d_out + outOff[s];
    k_gemm_wmma<2><<<dim3(cdiv(d, 128), cdiv(L, 64), B), 256, 0, stream>>>(
        YB, Wo, outChunk, (int)L, d, di, L * di, 0L, (long)d * L, nullptr, nullptr, (int)L);

    stageIn = F2;  // next stage consumes the conv2 feature map
  }
}
